// CDCNet_13606456394420
// MI455X (gfx1250) — matrix operand order repaired and checked by reading, not compiled, not run on hardware
//
#include <hip/hip_runtime.h>
#include <math.h>

#define NGR   128            // graphs
#define NPG   512            // nodes per graph
#define NTOT  (NGR * NPG)    // 65536 nodes
#define KNN   10
#define BNEPS 1e-5f

typedef __attribute__((ext_vector_type(16))) _Float16 v16h;
typedef __attribute__((ext_vector_type(8)))  _Float16 v8h;
typedef __attribute__((ext_vector_type(8)))  float    v8f;

// ---------------------------------------------------------------- utilities
__global__ void k_zero(float* p, int n) {
  for (int i = blockIdx.x * blockDim.x + threadIdx.x; i < n;
       i += gridDim.x * blockDim.x)
    p[i] = 0.f;
}

// column sum + sumsq via atomics into acc[0..C-1] (sum) / acc[C..2C-1] (sumsq)
__global__ void k_colstats(const float* __restrict__ x, int nrows, int C,
                           float* __restrict__ acc) {
  int c = blockIdx.x;
  int start = blockIdx.y * blockDim.x + threadIdx.x;
  int stride = gridDim.y * blockDim.x;
  float s = 0.f, q = 0.f;
  for (int r = start; r < nrows; r += stride) {
    float v = x[(size_t)r * C + c];
    s += v;
    q += v * v;
  }
  atomicAdd(&acc[c], s);
  atomicAdd(&acc[C + c], q);
}

// training-mode batchnorm apply (biased variance)
__global__ void k_bn_apply(const float* __restrict__ in, float* __restrict__ out,
                           int nrows, int C, const float* __restrict__ acc,
                           const float* __restrict__ g, const float* __restrict__ b) {
  size_t tot = (size_t)nrows * C;
  float invn = 1.f / (float)nrows;
  for (size_t i = blockIdx.x * (size_t)blockDim.x + threadIdx.x; i < tot;
       i += (size_t)gridDim.x * blockDim.x) {
    int c = (int)(i % C);
    float mu  = acc[c] * invn;
    float var = acc[C + c] * invn - mu * mu;
    out[i] = (in[i] - mu) * rsqrtf(var + BNEPS) * g[c] + b[c];
  }
}

// per-graph mean pool: pooled[g,c] = mean over NPG rows
__global__ void k_pool(const float* __restrict__ x, int C,
                       float* __restrict__ pooled) {
  int g = blockIdx.x;
  for (int c = threadIdx.x; c < C; c += blockDim.x) {
    float s = 0.f;
    for (int r = 0; r < NPG; ++r) s += x[((size_t)g * NPG + r) * C + c];
    pooled[g * C + c] = s * (1.f / NPG);
  }
}

// ---------------------------------------------------------------- packing
// A -> f16 [NTOT x Kpad], zero padded; handles 2-source concat + pooled bcast.
// grid.x = NTOT (one block per row), blockDim.x = Kpad (<= 256, mult of 32)
__global__ void k_packA(const float* __restrict__ A0, int ld0,
                        const float* __restrict__ A1, int ld1, int split,
                        int poolShift, int fin, _Float16* __restrict__ out) {
  int m = blockIdx.x;
  int k = threadIdx.x;
  float v = 0.f;
  if (k < split) {
    v = A0[(size_t)m * ld0 + k];
  } else if (k < fin) {
    int r = (poolShift >= 0) ? (m >> poolShift) : m;
    v = A1[(size_t)r * ld1 + (k - split)];
  }
  out[(size_t)m * blockDim.x + k] = (_Float16)v;
}

// W[k,n] (row-major fin x fout) -> f16 transposed+padded [Npad x Kpad]
// grid.x = Npad, blockDim.x = Kpad
__global__ void k_packW(const float* __restrict__ W, int fin, int fout,
                        _Float16* __restrict__ out) {
  int n = blockIdx.x;
  int k = threadIdx.x;
  float v = (k < fin && n < fout) ? W[(size_t)k * fout + n] : 0.f;
  out[(size_t)n * blockDim.x + k] = (_Float16)v;
}

// ---------------------------------------------------------------- WMMA GEMM
// D[m,n] = act( sum_k Ap[m,k] * Bp[n,k] + bias[n] )
// Ap: f16 [NTOT x Kpad] row-major; Bp: f16 [Npad x Kpad] (i.e. W transposed).
// One wave computes one 16x16 C tile; branch-free K loop of vector loads + wmma.
__global__ void __launch_bounds__(128)
k_gemm(const _Float16* __restrict__ Ap, int Kpad,
       const _Float16* __restrict__ Bp, const float* __restrict__ bias,
       float* __restrict__ D, int ldd, int Nout, int act) {
  const int lane  = threadIdx.x & 31;
  const int wave  = threadIdx.x >> 5;
  const int tileM = (blockIdx.x * 4 + wave) << 4;
  const int tileN = blockIdx.y << 4;
  const int hi    = (lane >> 4) & 1;   // 0: lanes 0-15, 1: lanes 16-31
  const int l4    = lane & 15;

  // A fragment lane source: row tileM+l4, K offsets k0+hi*16 (8 f16) and +8
  const _Float16* arow = Ap + (size_t)(tileM + l4) * Kpad + (hi << 4);
  // B fragment lane source: row tileN+l4 of transposed W, K offset k0+hi*16
  const _Float16* brow = Bp + (size_t)(tileN + l4) * Kpad + (hi << 4);

  v8f acc = {0.f, 0.f, 0.f, 0.f, 0.f, 0.f, 0.f, 0.f};
  for (int k0 = 0; k0 < Kpad; k0 += 32) {
    v8h alo = *(const v8h*)(arow + k0);        // K = k0+hi*16  .. +7
    v8h ahi = *(const v8h*)(arow + k0 + 8);    // K = k0+hi*16+8.. +15
    v16h af = __builtin_shufflevector(alo, ahi, 0, 1, 2, 3, 4, 5, 6, 7,
                                      8, 9, 10, 11, 12, 13, 14, 15);
    v16h bf = *(const v16h*)(brow + k0);       // K = k0+hi*16 .. +15
    acc = __builtin_amdgcn_wmma_f32_16x16x32_f16(
        false, af, false, bf, (short)0, acc, false, false);
  }

  int n = tileN + l4;
  if (n < Nout) {
    float bv = bias ? bias[n] : 0.f;
#pragma unroll
    for (int r = 0; r < 8; ++r) {
      int m = tileM + r + (hi << 3);
      float v = acc[r] + bv;
      if (act == 1) v = (v > 0.f) ? v : (expf(v) - 1.f);   // ELU
      D[(size_t)m * ldd + n] = v;
    }
  }
}

// ---------------------------------------------------------------- kNN (per graph)
__global__ void __launch_bounds__(256)
k_knn(const float* __restrict__ S, int* __restrict__ idx,
      float* __restrict__ wgt) {
  __shared__ float ss[NPG * 4];
  int g = blockIdx.x >> 1;
  int half = blockIdx.x & 1;
  int base = g * NPG;
  for (int t = threadIdx.x; t < NPG; t += 256) {
#pragma unroll
    for (int c = 0; c < 4; ++c) ss[t * 4 + c] = S[((size_t)base + t) * 4 + c];
  }
  __syncthreads();

  int nl = half * 256 + threadIdx.x;
  float s0 = ss[nl * 4 + 0], s1 = ss[nl * 4 + 1];
  float s2 = ss[nl * 4 + 2], s3 = ss[nl * 4 + 3];

  float bd[KNN];
  int   bi[KNN];
#pragma unroll
  for (int k = 0; k < KNN; ++k) { bd[k] = 3.4e38f; bi[k] = 0; }

  for (int j = 0; j < NPG; ++j) {
    float d0 = ss[j * 4 + 0] - s0, d1 = ss[j * 4 + 1] - s1;
    float d2 = ss[j * 4 + 2] - s2, d3 = ss[j * 4 + 3] - s3;
    float dist = d0 * d0 + d1 * d1 + d2 * d2 + d3 * d3;
    float v = dist;
    int   id = j;
#pragma unroll
    for (int k = 0; k < KNN; ++k) {       // sorted insert (ascending)
      if (v < bd[k]) {
        float tv = bd[k]; int ti = bi[k];
        bd[k] = v; bi[k] = id;
        v = tv; id = ti;
      }
    }
  }
  int n = base + nl;
#pragma unroll
  for (int k = 0; k < KNN; ++k) {
    idx[(size_t)n * KNN + k] = base + bi[k];
    wgt[(size_t)n * KNN + k] = expf(-10.f * bd[k]);
  }
}

// ---------------------------------------------------------------- aggregate
__global__ void k_aggregate(const float* __restrict__ h,
                            const int* __restrict__ idx,
                            const float* __restrict__ w,
                            float* __restrict__ agg) {
  size_t t = blockIdx.x * (size_t)blockDim.x + threadIdx.x;
  if (t >= (size_t)NTOT * 64) return;
  int n = (int)(t >> 6);
  int f = (int)(t & 63);
  float s = 0.f, mx = -3.4e38f;
#pragma unroll
  for (int k = 0; k < KNN; ++k) {
    int j = idx[(size_t)n * KNN + k];
    float v = h[(size_t)j * 64 + f] * w[(size_t)n * KNN + k];
    s += v;
    mx = fmaxf(mx, v);
  }
  agg[(size_t)n * 128 + f]      = s * (1.f / KNN);
  agg[(size_t)n * 128 + 64 + f] = mx;
}

// ---------------------------------------------------------------- output heads
__global__ void k_heads(const float* __restrict__ y,
                        const float* Wb, const float* bb,
                        const float* Wc, const float* bc,
                        const float* Wp, const float* bp,
                        const float* Wv, const float* bv,
                        const float* Wq, const float* bq,
                        float* __restrict__ out) {
  int n = blockIdx.x * blockDim.x + threadIdx.x;
  if (n >= NTOT) return;
  float a0 = bb[0];
  float a1 = bc[0], a2 = bc[1];
  float a3 = bp[0], a4 = bp[1], a5 = bp[2];
  float a6 = bv[0], a7 = bv[1], a8 = bv[2];
  float a9 = bq[0];
  for (int i = 0; i < 64; ++i) {
    float v = y[(size_t)n * 64 + i];
    a0 += v * Wb[i];
    a1 += v * Wc[i * 2 + 0]; a2 += v * Wc[i * 2 + 1];
    a3 += v * Wp[i * 3 + 0]; a4 += v * Wp[i * 3 + 1]; a5 += v * Wp[i * 3 + 2];
    a6 += v * Wv[i * 3 + 0]; a7 += v * Wv[i * 3 + 1]; a8 += v * Wv[i * 3 + 2];
    a9 += v * Wq[i];
  }
  float* o = out + (size_t)n * 10;
  o[0] = 1.f / (1.f + expf(-a0));
  o[1] = a1; o[2] = a2;
  o[3] = a3; o[4] = a4; o[5] = a5;
  o[6] = a6; o[7] = a7; o[8] = a8;
  o[9] = 1.f / (1.f + expf(-a9));
}

// ---------------------------------------------------------------- driver
extern "C" void kernel_launch(void* const* d_in, const int* in_sizes, int n_in,
                              void* d_out, int out_size, void* d_ws, size_t ws_size,
                              hipStream_t stream) {
  const float* x     = (const float*)d_in[0];
  // d_in[1] = batch (implicit: equal-size contiguous graphs)
  const float* bn0_g = (const float*)d_in[2];
  const float* bn0_b = (const float*)d_in[3];
  auto BP = [&](int blk, int item) -> const float* {
    return (const float*)d_in[4 + blk * 18 + item];
  };
  const float* dcW = (const float*)d_in[76];
  const float* dcB = (const float*)d_in[77];

  // ---- workspace carve-up
  float* f = (float*)d_ws;
  float* XN    = f; f += (size_t)NTOT * 6;      // bn0(x)
  float* POOL0 = f; f += (size_t)NGR * 6;
  float* XP    = f; f += (size_t)NTOT * 128;    // block output
  float* POOLP = f; f += (size_t)NGR * 128;
  float* T1    = f; f += (size_t)NTOT * 64;
  float* T2    = f; f += (size_t)NTOT * 64;
  float* XB    = f; f += (size_t)NTOT * 64;     // lin3 out
  float* Hh    = f; f += (size_t)NTOT * 64;     // gravnet h
  float* Sb    = f; f += (size_t)NTOT * 4;      // gravnet s
  float* AGG   = f; f += (size_t)NTOT * 128;
  float* Yf    = f; f += (size_t)NTOT * 128;    // concat feats
  float* WGT   = f; f += (size_t)NTOT * KNN;
  float* STATS = f; f += 256;
  int*       IDX   = (int*)f;                             // NTOT*KNN ints
  _Float16*  APACK = (_Float16*)(IDX + (size_t)NTOT * KNN); // NTOT*256 f16
  _Float16*  BPACK = APACK + (size_t)NTOT * 256;            // 128*256 f16

  // pack + gemm:  D = act(concat(A0,A1) @ W + bias)
  auto gemmL = [&](const float* A0, int ld0, const float* A1, int ld1,
                   int split, int poolShift, int fin, int Kpad,
                   const float* W, const float* bias, float* D, int ldd,
                   int fout, int act) {
    int Npad = (fout + 15) & ~15;
    k_packA<<<NTOT, Kpad, 0, stream>>>(A0, ld0, A1, ld1, split, poolShift,
                                       fin, APACK);
    k_packW<<<Npad, Kpad, 0, stream>>>(W, fin, fout, BPACK);
    dim3 grid(NTOT / 64, Npad / 16);
    k_gemm<<<grid, 128, 0, stream>>>(APACK, Kpad, BPACK, bias, D, ldd, fout, act);
  };
  auto bnorm = [&](const float* in, float* outp, int C, const float* g,
                   const float* b) {
    k_zero<<<1, 256, 0, stream>>>(STATS, 2 * C);
    dim3 gs(C, 32);
    k_colstats<<<gs, 256, 0, stream>>>(in, NTOT, C, STATS);
    k_bn_apply<<<4096, 256, 0, stream>>>(in, outp, NTOT, C, STATS, g, b);
  };

  // ---- input BN + initial pool
  bnorm(x, XN, 6, bn0_g, bn0_b);
  k_pool<<<NGR, 128, 0, stream>>>(XN, 6, POOL0);

  for (int i = 0; i < 4; ++i) {
    const float* A0;  int ld0, fin, split, Kpad;
    const float* A1;  int ld1;
    if (i == 0) {
      A0 = XN; ld0 = 6; A1 = POOL0; ld1 = 6; split = 6; fin = 12; Kpad = 32;
    } else {
      k_pool<<<NGR, 128, 0, stream>>>(XP, 128, POOLP);
      A0 = XP; ld0 = 128; A1 = POOLP; ld1 = 128; split = 128; fin = 256;
      Kpad = 256;
    }
    // lin0 (ELU) -> T1 ; lin1 (ELU) -> T2 ; bn_a in place
    gemmL(A0, ld0, A1, ld1, split, 9, fin, Kpad, BP(i,0), BP(i,1), T1, 64, 64, 1);
    gemmL(T1, 64, T1, 64, 64, -1, 64, 64, BP(i,2), BP(i,3), T2, 64, 64, 1);
    bnorm(T2, T2, 64, BP(i,4), BP(i,5));
    // lin3 (ELU) -> XB
    gemmL(T2, 64, T2, 64, 64, -1, 64, 64, BP(i,6), BP(i,7), XB, 64, 64, 1);
    // gravnet: s, h, knn, aggregate, gn_o
    gemmL(XB, 64, XB, 64, 64, -1, 64, 64, BP(i,8),  BP(i,9),  Sb, 4,  4,  0);
    gemmL(XB, 64, XB, 64, 64, -1, 64, 64, BP(i,10), BP(i,11), Hh, 64, 64, 0);
    k_knn<<<NGR * 2, 256, 0, stream>>>(Sb, IDX, WGT);
    k_aggregate<<<(NTOT * 64) / 256, 256, 0, stream>>>(Hh, IDX, WGT, AGG);
    gemmL(XB, 64, AGG, 128, 64, -1, 192, 192, BP(i,12), BP(i,13), XP, 128,
          128, 0);
    // bn_b in place, lin6 (ELU) into feature slab
    bnorm(XP, XP, 128, BP(i,14), BP(i,15));
    gemmL(XP, 128, XP, 128, 128, -1, 128, 128, BP(i,16), BP(i,17),
          Yf + 32 * i, 128, 32, 1);
  }

  // dense_cat (ELU) -> T1 (reused as [N,64])
  gemmL(Yf, 128, Yf, 128, 128, -1, 128, 128, dcW, dcB, T1, 64, 64, 1);

  // heads
  k_heads<<<(NTOT + 255) / 256, 256, 0, stream>>>(
      T1,
      (const float*)d_in[78], (const float*)d_in[79],
      (const float*)d_in[80], (const float*)d_in[81],
      (const float*)d_in[82], (const float*)d_in[83],
      (const float*)d_in[84], (const float*)d_in[85],
      (const float*)d_in[86], (const float*)d_in[87],
      (float*)d_out);
}
